// MetaNetImageEncoder_57483842289849
// MI455X (gfx1250) — matrix operand order repaired and checked by reading, not compiled, not run on hardware
//
#include <hip/hip_runtime.h>

// ---------------------------------------------------------------------------
// MetaNet image encoder for MI455X (gfx1250, wave32, WMMA bf16 16x16x32).
//
// Shapes: B=64, N=196, D=768, T=8, H_META=192. Rows of the big GEMMs:
// B*N = 12544 (= 784 tiles of 16 exactly).
// ---------------------------------------------------------------------------

typedef __bf16 bf16_t;
typedef __attribute__((ext_vector_type(16))) __bf16 v16bf;
typedef __attribute__((ext_vector_type(8)))  __bf16 v8bf;
typedef __attribute__((ext_vector_type(8)))  float  v8f;

#define D_DIM   768
#define N_PATCH 196
#define B_SZ    64
#define T_SZ    8
#define H_META  192
#define ROWS    (B_SZ * N_PATCH)   // 12544
#define DXD     (D_DIM * D_DIM)    // 589824

__device__ inline v8f vzero8() {
  v8f v;
#pragma unroll
  for (int i = 0; i < 8; ++i) v[i] = 0.0f;
  return v;
}

__device__ inline v16bf make_frag(v8bf lo, v8bf hi) {
  v16bf r;
#pragma unroll
  for (int i = 0; i < 8; ++i) { r[i] = lo[i]; r[i + 8] = hi[i]; }
  return r;
}

// A fragment (16x32 bf16, row-major source, per-lane row fixed):
// elements 0..7  = K [kA0   .. kA0+7 ]   (contiguous)
// elements 8..15 = K [kA0+8 .. kA0+15]   (contiguous)
// caller passes p already offset by kA0; second run is +8 elements.
__device__ inline v16bf load_frag_a(const bf16_t* p) {
  v8bf lo = *(const v8bf*)p;
  v8bf hi = *(const v8bf*)(p + 8);
  return make_frag(lo, hi);
}

// B fragment (32x16 bf16) from transposed [e][d] storage: per-lane column
// fixed, 16 contiguous K values starting at kB0 (0 or 16).
__device__ inline v16bf load_frag_b(const bf16_t* p) {
  v8bf lo = *(const v8bf*)p;
  v8bf hi = *(const v8bf*)(p + 8);
  return make_frag(lo, hi);
}

__device__ inline v8f wmma_bf16(v16bf a, v16bf b, v8f c) {
  return __builtin_amdgcn_wmma_f32_16x16x32_bf16(
      /*neg_a=*/false, a, /*neg_b=*/false, b,
      /*c_mod=*/(short)0, c, /*reuse_a=*/false, /*reuse_b=*/false);
}

// ---------------------------------------------------------------------------
// K1: patchify x[B,3,224,224] f32 -> P[B*N, D] bf16.  idx is output-linear.
// ---------------------------------------------------------------------------
__global__ __launch_bounds__(256) void patchify_kernel(
    const float* __restrict__ x, bf16_t* __restrict__ P) {
  int idx = blockIdx.x * 256 + threadIdx.x;           // < 9,633,792 (exact)
  int d = idx % D_DIM;
  int r = idx / D_DIM;
  int p = r % N_PATCH;
  int b = r / N_PATCH;
  int c  = d >> 8;
  int ph = (d >> 4) & 15;
  int pw = d & 15;
  int hp = p / 14, wp = p % 14;
  float v = x[((b * 3 + c) * 224 + hp * 16 + ph) * 224 + wp * 16 + pw];
  P[idx] = (bf16_t)v;
}

// ---------------------------------------------------------------------------
// K2: transpose+convert nmat matrices [d][e] f32 -> [e][d] bf16.
// ---------------------------------------------------------------------------
__global__ __launch_bounds__(256) void transpose_bf16_kernel(
    const float* __restrict__ src, bf16_t* __restrict__ dst, int nmat) {
  int idx = blockIdx.x * 256 + threadIdx.x;
  int total = nmat * DXD;
  if (idx >= total) return;
  int m = idx / DXD;
  int r = idx % DXD;
  int e = r / D_DIM, d = r % D_DIM;
  dst[idx] = (bf16_t)src[m * DXD + d * D_DIM + e];
}

// ---------------------------------------------------------------------------
// K3: base GEMM  H = relu(P[12544x768] @ W1 + b1), W1 given transposed bf16.
// Grid (49 row-supertiles of 256, 24 col-tiles of 32); 8 waves; each wave
// does a 2x2 register block: 2 row tiles x 2 col subtiles -> 4 WMMAs per
// K step with full A/B fragment reuse (2 b128 loads per WMMA).
// ---------------------------------------------------------------------------
__global__ __launch_bounds__(256) void gemm_base_kernel(
    const bf16_t* __restrict__ A, const bf16_t* __restrict__ BT,
    const float* __restrict__ bias, bf16_t* __restrict__ H) {
  const int tid  = threadIdx.x;
  const int wave = tid >> 5, lane = tid & 31;
  const int laneM = lane & 15;
  const bool hi   = lane >= 16;
  const int row0 = blockIdx.x * 256 + wave * 32;   // two 16-row tiles
  const int col0 = blockIdx.y * 32;
  const int kA0 = hi ? 16 : 0;
  const int kB0 = hi ? 16 : 0;

  const bf16_t* Ap0 = A  + (size_t)(row0 + laneM) * D_DIM + kA0;
  const bf16_t* Ap1 = Ap0 + (size_t)16 * D_DIM;
  const bf16_t* Bp0 = BT + (size_t)(col0 + laneM) * D_DIM + kB0;
  const bf16_t* Bp1 = BT + (size_t)(col0 + 16 + laneM) * D_DIM + kB0;

  v8f acc00 = vzero8(), acc01 = vzero8(), acc10 = vzero8(), acc11 = vzero8();
#pragma unroll 2
  for (int k = 0; k < D_DIM; k += 32) {
    v16bf fa0 = load_frag_a(Ap0 + k);
    v16bf fa1 = load_frag_a(Ap1 + k);
    v16bf fb0 = load_frag_b(Bp0 + k);
    v16bf fb1 = load_frag_b(Bp1 + k);
    acc00 = wmma_bf16(fa0, fb0, acc00);
    acc01 = wmma_bf16(fa0, fb1, acc01);
    acc10 = wmma_bf16(fa1, fb0, acc10);
    acc11 = wmma_bf16(fa1, fb1, acc11);
  }
  const float bz0 = bias[col0 + laneM];
  const float bz1 = bias[col0 + 16 + laneM];
  bf16_t* Hp0 = H + (size_t)(row0 + (hi ? 8 : 0)) * D_DIM + col0 + laneM;
  bf16_t* Hp1 = Hp0 + (size_t)16 * D_DIM;
#pragma unroll
  for (int i = 0; i < 8; ++i) {
    float h00 = acc00[i] + bz0; h00 = h00 > 0.0f ? h00 : 0.0f;
    float h01 = acc01[i] + bz1; h01 = h01 > 0.0f ? h01 : 0.0f;
    float h10 = acc10[i] + bz0; h10 = h10 > 0.0f ? h10 : 0.0f;
    float h11 = acc11[i] + bz1; h11 = h11 > 0.0f ? h11 : 0.0f;
    Hp0[(size_t)i * D_DIM]      = (bf16_t)h00;
    Hp0[(size_t)i * D_DIM + 16] = (bf16_t)h01;
    Hp1[(size_t)i * D_DIM]      = (bf16_t)h10;
    Hp1[(size_t)i * D_DIM + 16] = (bf16_t)h11;
  }
}

// ---------------------------------------------------------------------------
// K4: mean-pool H[b, 0..195, e] -> pooled f32 + bf16 copies. Grid (3, 64).
// ---------------------------------------------------------------------------
__global__ __launch_bounds__(256) void pool_kernel(
    const bf16_t* __restrict__ H, float* __restrict__ pf,
    bf16_t* __restrict__ pb) {
  int b = blockIdx.y;
  int e = blockIdx.x * 256 + threadIdx.x;
  const bf16_t* hp = H + (size_t)b * N_PATCH * D_DIM + e;
  float s = 0.0f;
  for (int n = 0; n < N_PATCH; ++n) s += (float)hp[(size_t)n * D_DIM];
  s *= (1.0f / (float)N_PATCH);
  pf[b * D_DIM + e] = s;
  pb[b * D_DIM + e] = (bf16_t)s;
}

// ---------------------------------------------------------------------------
// K5: fused base head + MetaNet. One block per sample.
//   base = pooled @ W2 + b2 ; hid = relu(base @ mw1 + mb1) ;
//   coefs = hid @ mw2 + mb2
// ---------------------------------------------------------------------------
__global__ __launch_bounds__(256) void metanet_kernel(
    const float* __restrict__ pooled, const float* __restrict__ W2,
    const float* __restrict__ b2, const float* __restrict__ mw1,
    const float* __restrict__ mb1, const float* __restrict__ mw2,
    const float* __restrict__ mb2, float* __restrict__ coefs) {
  __shared__ float sp[D_DIM];
  __shared__ float sbase[D_DIM];
  __shared__ float shid[H_META];
  const int b = blockIdx.x, tid = threadIdx.x;
  for (int e = tid; e < D_DIM; e += 256) sp[e] = pooled[b * D_DIM + e];
  __syncthreads();
  for (int e = tid; e < D_DIM; e += 256) {
    float acc = b2[e];
    for (int d = 0; d < D_DIM; ++d) acc += sp[d] * W2[d * D_DIM + e];
    sbase[e] = acc;
  }
  __syncthreads();
  if (tid < H_META) {
    float acc = mb1[tid];
    for (int d = 0; d < D_DIM; ++d) acc += sbase[d] * mw1[d * H_META + tid];
    shid[tid] = acc > 0.0f ? acc : 0.0f;
  }
  __syncthreads();
  if (tid < T_SZ) {
    float acc = mb2[tid];
    for (int h = 0; h < H_META; ++h) acc += shid[h] * mw2[h * T_SZ + tid];
    coefs[b * T_SZ + tid] = acc;
  }
}

// ---------------------------------------------------------------------------
// K6: mixed per-sample GEMM.  For block (col-tile e0..e0+31, sample b):
//   stage Wmix[e][d] = W1[d][e] + sum_t c[b,t]*dW1[t][d][e] into LDS (bf16,
//   transposed, padded stride 776 to dodge bank conflicts), then WMMA
//   H2[b,n,:] = relu(P[b,n,:] @ Wmix + nb1).  dW1 (18.9 MB) stays in L2.
// Waves process row-tile PAIRS so each LDS B fragment feeds 2 WMMAs.
// 13 row tiles -> 7 pairs (last pair duplicates tile 12; stores idempotent).
// ---------------------------------------------------------------------------
__global__ __launch_bounds__(256) void gemm_mixed_kernel(
    const bf16_t* __restrict__ P, const float* __restrict__ W1,
    const float* __restrict__ dW1, const float* __restrict__ b1,
    const float* __restrict__ db1, const float* __restrict__ coefs,
    bf16_t* __restrict__ H) {
  __shared__ bf16_t wmix[32][776];   // [e_local][d], 49.7 KB
  __shared__ float sc[T_SZ];
  __shared__ float sbias[32];
  const int tid  = threadIdx.x;
  const int b    = blockIdx.y;
  const int col0 = blockIdx.x * 32;

  if (tid < T_SZ) sc[tid] = coefs[b * T_SZ + tid];
  __syncthreads();

  { // cooperative mixed-weight staging: 32 e-lanes x 8 d-lanes per pass
    const int e = tid & 31;
    for (int d = tid >> 5; d < D_DIM; d += 8) {
      float w = W1[d * D_DIM + col0 + e];
#pragma unroll
      for (int t = 0; t < T_SZ; ++t)
        w += sc[t] * dW1[(t * D_DIM + d) * D_DIM + col0 + e];
      wmix[e][d] = (bf16_t)w;
    }
  }
  if (tid < 32) {
    float bv = b1[col0 + tid];
#pragma unroll
    for (int t = 0; t < T_SZ; ++t) bv += sc[t] * db1[t * D_DIM + col0 + tid];
    sbias[tid] = bv;
  }
  __syncthreads();

  const int wave = tid >> 5, lane = tid & 31;
  const int laneM = lane & 15;
  const bool hi   = lane >= 16;
  const int kA0 = hi ? 16 : 0;
  const int kB0 = hi ? 16 : 0;
  const bf16_t* Bls0 = &wmix[laneM][kB0];
  const bf16_t* Bls1 = &wmix[16 + laneM][kB0];
  const float bz0 = sbias[laneM];
  const float bz1 = sbias[16 + laneM];

  for (int rp = wave; rp * 2 < 13; rp += 8) {  // 7 pairs; wave 7 idle
    const int rt0 = rp * 2;
    const int rt1 = (rp * 2 + 1 < 13) ? rp * 2 + 1 : 12;  // dup tile at tail
    int an0 = rt0 * 16 + laneM; if (an0 > N_PATCH - 1) an0 = N_PATCH - 1;
    int an1 = rt1 * 16 + laneM; if (an1 > N_PATCH - 1) an1 = N_PATCH - 1;
    const bf16_t* Ap0 = P + (size_t)(b * N_PATCH + an0) * D_DIM + kA0;
    const bf16_t* Ap1 = P + (size_t)(b * N_PATCH + an1) * D_DIM + kA0;
    v8f acc00 = vzero8(), acc01 = vzero8(), acc10 = vzero8(), acc11 = vzero8();
#pragma unroll 2
    for (int k = 0; k < D_DIM; k += 32) {
      v16bf fa0 = load_frag_a(Ap0 + k);
      v16bf fa1 = load_frag_a(Ap1 + k);
      v16bf fb0 = load_frag_b(Bls0 + k);
      v16bf fb1 = load_frag_b(Bls1 + k);
      acc00 = wmma_bf16(fa0, fb0, acc00);
      acc01 = wmma_bf16(fa0, fb1, acc01);
      acc10 = wmma_bf16(fa1, fb0, acc10);
      acc11 = wmma_bf16(fa1, fb1, acc11);
    }
#pragma unroll
    for (int i = 0; i < 8; ++i) {
      int n0 = rt0 * 16 + i + (hi ? 8 : 0);
      int n1 = rt1 * 16 + i + (hi ? 8 : 0);
      if (n0 < N_PATCH) {
        float h0 = acc00[i] + bz0; h0 = h0 > 0.0f ? h0 : 0.0f;
        float h1 = acc01[i] + bz1; h1 = h1 > 0.0f ? h1 : 0.0f;
        size_t base = (size_t)(b * N_PATCH + n0) * D_DIM + col0 + laneM;
        H[base]      = (bf16_t)h0;
        H[base + 16] = (bf16_t)h1;
      }
      if (n1 < N_PATCH) {
        float h0 = acc10[i] + bz0; h0 = h0 > 0.0f ? h0 : 0.0f;
        float h1 = acc11[i] + bz1; h1 = h1 > 0.0f ? h1 : 0.0f;
        size_t base = (size_t)(b * N_PATCH + n1) * D_DIM + col0 + laneM;
        H[base]      = (bf16_t)h0;
        H[base + 16] = (bf16_t)h1;
      }
    }
  }
}

// ---------------------------------------------------------------------------
// K7: head GEMMs.  G[s] = pooled2_bf @ (s==0 ? W2 : dW2[s-1]),  s = 0..8.
// Grid (24 col-tiles, 9 s); wave -> (row tile 0..3, col subtile 0..1).
// ---------------------------------------------------------------------------
__global__ __launch_bounds__(256) void gemm_head_kernel(
    const bf16_t* __restrict__ A, const bf16_t* __restrict__ W2T,
    const bf16_t* __restrict__ dW2T, float* __restrict__ G) {
  const int tid  = threadIdx.x;
  const int wave = tid >> 5, lane = tid & 31;
  const int laneM = lane & 15;
  const bool hi   = lane >= 16;
  const int s = blockIdx.y;
  const bf16_t* BT = (s == 0) ? W2T : (dW2T + (size_t)(s - 1) * DXD);
  const int row0 = (wave >> 1) * 16;                 // 0,16,32,48
  const int col  = blockIdx.x * 32 + (wave & 1) * 16 + laneM;
  const int kA0 = hi ? 16 : 0;
  const int kB0 = hi ? 16 : 0;
  const bf16_t* Ap = A  + (row0 + laneM) * D_DIM + kA0;
  const bf16_t* Bp = BT + (size_t)col * D_DIM + kB0;
  v8f acc = vzero8();
#pragma unroll 4
  for (int k = 0; k < D_DIM; k += 32)
    acc = wmma_bf16(load_frag_a(Ap + k), load_frag_b(Bp + k), acc);
  float* Gp = G + (size_t)(s * B_SZ + row0 + (hi ? 8 : 0)) * D_DIM + col;
#pragma unroll
  for (int i = 0; i < 8; ++i) Gp[(size_t)i * D_DIM] = acc[i];
}

// ---------------------------------------------------------------------------
// K8: out[b,e] = G[0][b,e] + b2[e] + sum_t c[b,t]*(G[1+t][b,e] + db2[t,e])
// ---------------------------------------------------------------------------
__global__ __launch_bounds__(256) void combine_kernel(
    const float* __restrict__ G, const float* __restrict__ coefs,
    const float* __restrict__ b2, const float* __restrict__ db2,
    float* __restrict__ out) {
  int idx = blockIdx.x * 256 + threadIdx.x;          // < 49152 (exact)
  int b = idx / D_DIM, e = idx % D_DIM;
  float v = G[(size_t)b * D_DIM + e] + b2[e];
#pragma unroll
  for (int t = 0; t < T_SZ; ++t) {
    float c = coefs[b * T_SZ + t];
    v += c * (G[(size_t)((1 + t) * B_SZ + b) * D_DIM + e] + db2[t * D_DIM + e]);
  }
  out[idx] = v;
}

// ---------------------------------------------------------------------------
// Host launcher
// ---------------------------------------------------------------------------
extern "C" void kernel_launch(void* const* d_in, const int* in_sizes, int n_in,
                              void* d_out, int out_size, void* d_ws,
                              size_t ws_size, hipStream_t stream) {
  (void)in_sizes; (void)n_in; (void)out_size; (void)ws_size;
  const float* x   = (const float*)d_in[0];
  const float* W1  = (const float*)d_in[1];
  const float* b1  = (const float*)d_in[2];
  const float* W2  = (const float*)d_in[3];
  const float* b2  = (const float*)d_in[4];
  const float* dW1 = (const float*)d_in[5];
  const float* db1 = (const float*)d_in[6];
  const float* dW2 = (const float*)d_in[7];
  const float* db2 = (const float*)d_in[8];
  const float* mw1 = (const float*)d_in[9];
  const float* mb1 = (const float*)d_in[10];
  const float* mw2 = (const float*)d_in[11];
  const float* mb2 = (const float*)d_in[12];
  float* out = (float*)d_out;

  char* ws = (char*)d_ws;
  size_t off = 0;
  auto alloc = [&](size_t bytes) -> void* {
    void* p = ws + off;
    off = (off + bytes + 255) & ~(size_t)255;
    return p;
  };

  bf16_t* P        = (bf16_t*)alloc((size_t)ROWS * D_DIM * 2);       // 19.3 MB
  bf16_t* W1T      = (bf16_t*)alloc((size_t)DXD * 2);
  bf16_t* W2T      = (bf16_t*)alloc((size_t)DXD * 2);
  bf16_t* dW2T     = (bf16_t*)alloc((size_t)T_SZ * DXD * 2);         //  9.4 MB
  bf16_t* Hbuf     = (bf16_t*)alloc((size_t)ROWS * D_DIM * 2);       // 19.3 MB
  float*  pooled1  = (float*) alloc((size_t)B_SZ * D_DIM * 4);
  bf16_t* pooled1b = (bf16_t*)alloc((size_t)B_SZ * D_DIM * 2);
  float*  pooled2  = (float*) alloc((size_t)B_SZ * D_DIM * 4);
  bf16_t* pooled2b = (bf16_t*)alloc((size_t)B_SZ * D_DIM * 2);
  float*  coefs    = (float*) alloc((size_t)B_SZ * T_SZ * 4);
  float*  G        = (float*) alloc((size_t)9 * B_SZ * D_DIM * 4);   //  1.8 MB

  transpose_bf16_kernel<<<dim3(DXD / 256), 256, 0, stream>>>(W1, W1T, 1);
  transpose_bf16_kernel<<<dim3(DXD / 256), 256, 0, stream>>>(W2, W2T, 1);
  transpose_bf16_kernel<<<dim3(T_SZ * DXD / 256), 256, 0, stream>>>(dW2, dW2T, T_SZ);
  patchify_kernel<<<dim3((size_t)ROWS * D_DIM / 256), 256, 0, stream>>>(x, P);

  gemm_base_kernel<<<dim3(ROWS / 256, D_DIM / 32), 256, 0, stream>>>(P, W1T, b1, Hbuf);
  pool_kernel<<<dim3(D_DIM / 256, B_SZ), 256, 0, stream>>>(Hbuf, pooled1, pooled1b);
  metanet_kernel<<<dim3(B_SZ), 256, 0, stream>>>(pooled1, W2, b2, mw1, mb1, mw2, mb2, coefs);

  gemm_mixed_kernel<<<dim3(D_DIM / 32, B_SZ), 256, 0, stream>>>(P, W1, dW1, b1, db1, coefs, Hbuf);
  pool_kernel<<<dim3(D_DIM / 256, B_SZ), 256, 0, stream>>>(Hbuf, pooled2, pooled2b);

  gemm_head_kernel<<<dim3(D_DIM / 32, 9), 256, 0, stream>>>(pooled2b, W2T, dW2T, G);
  combine_kernel<<<dim3(B_SZ * D_DIM / 256), 256, 0, stream>>>(G, coefs, b2, db2, out);
}
